// HeteroGAT_43327630082180
// MI455X (gfx1250) — matrix operand order repaired and checked by reading, not compiled, not run on hardware
//
#include <hip/hip_runtime.h>
#include <hip/hip_bf16.h>
#include <cstdint>

// ---------------------------------------------------------------------------
// HeteroGAT on MI455X (gfx1250, wave32).
// GEMMs: v_wmma_f32_16x16x32_bf16 with the weight tile staged into LDS by the
// Tensor Data Mover (tensor_load_to_lds + s_wait_tensorcnt) so all 8 waves of
// a block share one DMA instead of 8x redundant L2 reads. Per k-step all four
// B fragments are loaded before the four WMMAs so the XDL ops issue
// back-to-back behind a single dscnt wait. Attention/softmax is a
// wave-per-node kernel (16 contiguous incoming edges per node) using wave32
// shfl_xor reductions and LDS-staged alpha. Working set fits in the 192MB L2.
// ---------------------------------------------------------------------------

constexpr int NN    = 30000;       // nodes
constexpr int DEG   = 16;          // in-degree per node per etype
constexpr int EE    = NN * DEG;    // edges per etype
constexpr int TT    = 3;           // edge types
constexpr int FIN   = 128;         // in_feats
constexpr int HIDN  = 256;         // hidden
constexpr int HEADS = 4;

typedef __attribute__((ext_vector_type(16))) __bf16 bf16x16;
typedef __attribute__((ext_vector_type(8)))  __bf16 bf16x8;
typedef __attribute__((ext_vector_type(8)))  float  v8f;
typedef __attribute__((ext_vector_type(4)))  unsigned int u32x4;
typedef __attribute__((ext_vector_type(8)))  int    i32x8;
typedef __attribute__((ext_vector_type(4)))  int    i32x4;

union Frag { bf16x16 v; bf16x8 h[2]; };

__device__ __forceinline__ float lrelu(float x) { return x > 0.f ? x : 0.2f * x; }

// ---------------------------------------------------------------------------
// f32 -> bf16 conversions / weight transpose ([T][K][256] -> [T][256][K])
// ---------------------------------------------------------------------------
__global__ void cvt_feat_kernel(const float* __restrict__ in, __bf16* __restrict__ out, int count) {
  int id = blockIdx.x * 256 + threadIdx.x;
  if (id < count) out[id] = (__bf16)in[id];
}

__global__ void cvt_wt_kernel(const float* __restrict__ W, __bf16* __restrict__ Wt, int K) {
  int id = blockIdx.x * 256 + threadIdx.x;
  int total = TT * K * HIDN;
  if (id >= total) return;
  int c = id % HIDN;
  int k = (id / HIDN) % K;
  int t = id / (HIDN * K);
  Wt[((size_t)t * HIDN + c) * K + k] = (__bf16)W[id];
}

// ---------------------------------------------------------------------------
// WMMA GEMM: C[t][N][256] = A[N][KK](bf16) x W[t][KK][256] (Bt = [t][256][KK] bf16)
// Block: 8 waves -> 128 rows x 64 cols. The 64xKK weight tile (16/32KB) is
// DMA'd into LDS once per block by the TDM; wave frags then come from LDS.
// A frag (16-bit 16x32): lane<16 holds K{0..7,16..23}, lane>=16 K{8..15,24..31}.
// B frag: same K order as A (lane<16: K{0..7,16..23}, lane>=16: K{8..15,24..31}).
// ---------------------------------------------------------------------------
template <int KK>
__global__ __launch_bounds__(256) void gemm_wmma(const __bf16* __restrict__ A,
                                                 const __bf16* __restrict__ Bt,
                                                 float* __restrict__ C) {
  __shared__ __attribute__((aligned(16))) __bf16 sB[64 * KK];
  const int lane = threadIdx.x & 31;
  const int wave = threadIdx.x >> 5;
  const int col0 = blockIdx.y * 64;
  const int t    = blockIdx.z;

  // --- TDM: DMA the 64 x KK bf16 weight tile (rows col0..col0+63 of Bt[t])
  // into LDS. One descriptor per block, issued by wave 0 (TDM ignores EXEC).
  if (wave == 0) {
    unsigned long long ga =
        (unsigned long long)(uintptr_t)(Bt + ((size_t)t * HIDN + col0) * KK);
    unsigned ldsOff = (unsigned)(uintptr_t)(&sB[0]);
    u32x4 g0;
    g0[0] = 1u;                                               // count=1, user D#
    g0[1] = ldsOff;                                           // lds_addr (bytes)
    g0[2] = (unsigned)(ga & 0xFFFFFFFFull);                   // global_addr lo
    g0[3] = (unsigned)((ga >> 32) & 0x01FFFFFFull) | (2u << 30); // addr hi | type=2
    i32x8 g1;
    g1[0] = 1 << 16;                    // workgroup_mask=0, data_size=1 (2 bytes)
    g1[1] = (KK & 0xFFFF) << 16;        // tensor_dim0[15:0]
    g1[2] = (KK >> 16) | (64 << 16);    // tensor_dim0[31:16] | tensor_dim1[15:0]=64
    g1[3] = (KK & 0xFFFF) << 16;        // tensor_dim1[31:16]=0 | tile_dim0=KK
    g1[4] = 64;                         // tile_dim1=64, tile_dim2=0
    g1[5] = KK;                         // tensor_dim0_stride[31:0]=KK
    g1[6] = 0;                          // stride hi / dim1_stride lo (2D: unused)
    g1[7] = 0;
    i32x4 g2 = {0, 0, 0, 0};
    i32x4 g3 = {0, 0, 0, 0};
    i32x8 g4 = {0, 0, 0, 0, 0, 0, 0, 0};   // extra group (6-arg toolchain form)
    __builtin_amdgcn_tensor_load_to_lds(g0, g1, g2, g3, g4, 0);
    __builtin_amdgcn_s_wait_tensorcnt(0);
  }
  __syncthreads();                       // publish LDS tile to all 8 waves

  const int rowTile = blockIdx.x * 8 + wave;
  if (rowTile * 16 >= NN) return;        // after the barrier: safe
  const int row0 = rowTile * 16;
  const int m    = lane & 15;
  const int aOff = (lane < 16) ? 0 : 8;
  const int bOff = (lane < 16) ? 0 : 8;

  v8f acc[4] = {};
  #pragma unroll
  for (int k = 0; k < KK; k += 32) {
    Frag a;
    const bf16x8* ap = (const bf16x8*)(A + (size_t)(row0 + m) * KK + k + aOff);
    a.h[0] = ap[0];
    a.h[1] = ap[2];       // +16 elements
    // Load all four B fragments first (distinct registers), then issue the
    // four WMMAs back-to-back behind a single LDS wait.
    Frag b[4];
    #pragma unroll
    for (int ct = 0; ct < 4; ++ct) {
      const bf16x8* bp = (const bf16x8*)(&sB[(size_t)(ct * 16 + m) * KK + k + bOff]);
      b[ct].h[0] = bp[0];
      b[ct].h[1] = bp[2];
    }
    #pragma unroll
    for (int ct = 0; ct < 4; ++ct) {
      acc[ct] = __builtin_amdgcn_wmma_f32_16x16x32_bf16(
          false, a.v, false, b[ct].v, (short)0, acc[ct], false, false);
    }
  }
  // C layout: lane -> N = lane&15, VGPR r -> M = r + 8*(lane>>4)
  float* Cout = C + ((size_t)t * NN + row0) * HIDN + col0;
  const int rbase = (lane >> 4) * 8;
  #pragma unroll
  for (int ct = 0; ct < 4; ++ct) {
    #pragma unroll
    for (int r = 0; r < 8; ++r)
      Cout[(size_t)(rbase + r) * HIDN + ct * 16 + (lane & 15)] = acc[ct][r];
  }
}

// ---------------------------------------------------------------------------
// el/er: [T][N][4] (stored float4/node) from hproj and al/ar [T][4][64]
// ---------------------------------------------------------------------------
__global__ __launch_bounds__(256) void scores_kernel(const float* __restrict__ hproj,
                                                     const float* __restrict__ al,
                                                     const float* __restrict__ ar,
                                                     float* __restrict__ el,
                                                     float* __restrict__ er) {
  int id = blockIdx.x * 256 + threadIdx.x;
  if (id >= TT * NN * HEADS) return;
  int t = id / (NN * HEADS);
  int rem = id % (NN * HEADS);
  int n = rem >> 2;
  int h = rem & 3;
  const float* row = hproj + ((size_t)t * NN + n) * HIDN + h * 64;
  const float* alv = al + ((size_t)t * HEADS + h) * 64;
  const float* arv = ar + ((size_t)t * HEADS + h) * 64;
  float accl = 0.f, accr = 0.f;
  #pragma unroll 8
  for (int d = 0; d < 64; ++d) {
    float hv = row[d];
    accl = fmaf(hv, alv[d], accl);
    accr = fmaf(hv, arv[d], accr);
  }
  el[((size_t)t * NN + n) * 4 + h] = accl;
  er[((size_t)t * NN + n) * 4 + h] = accr;
}

// ---------------------------------------------------------------------------
// Aggregation: wave per node. Softmax over the node's 16 contiguous edges
// (per head) via shfl_xor within 16-lane groups; alpha staged in LDS; then
// 16 weighted float4x2 row accumulations, summed over T, mean+bias+ELU.
// Grid must be exactly NN/8 blocks (NN % 8 == 0).
// ---------------------------------------------------------------------------
__global__ __launch_bounds__(256) void agg_kernel(const int* __restrict__ src,
                                                  const float* __restrict__ el,
                                                  const float* __restrict__ er,
                                                  const float* __restrict__ hproj,
                                                  const float* __restrict__ bias,
                                                  float* __restrict__ outf,
                                                  __bf16* __restrict__ outbf,
                                                  int apply_elu) {
  __shared__ float s_alpha[8][DEG * HEADS];
  __shared__ int   s_src[8][DEG];
  const int lane = threadIdx.x & 31;
  const int w    = threadIdx.x >> 5;
  const int node = blockIdx.x * 8 + w;
  const int j    = lane & 15;        // edge slot (duplicated in both lane halves)
  const int head = lane >> 3;        // 8 cols per lane -> head = col/64
  const int colb = lane * 8;
  float acc[8] = {0.f, 0.f, 0.f, 0.f, 0.f, 0.f, 0.f, 0.f};

  for (int t = 0; t < TT; ++t) {
    const int sidx = src[(size_t)t * EE + node * DEG + j];
    const float4 el4 = *(const float4*)(el + ((size_t)t * NN + sidx) * 4);
    const float4 er4 = *(const float4*)(er + ((size_t)t * NN + node) * 4);
    float4 e;
    e.x = lrelu(el4.x + er4.x);
    e.y = lrelu(el4.y + er4.y);
    e.z = lrelu(el4.z + er4.z);
    e.w = lrelu(el4.w + er4.w);
    float4 mx = e;
    #pragma unroll
    for (int msk = 1; msk < 16; msk <<= 1) {
      mx.x = fmaxf(mx.x, __shfl_xor(mx.x, msk, 32));
      mx.y = fmaxf(mx.y, __shfl_xor(mx.y, msk, 32));
      mx.z = fmaxf(mx.z, __shfl_xor(mx.z, msk, 32));
      mx.w = fmaxf(mx.w, __shfl_xor(mx.w, msk, 32));
    }
    float4 ex;
    ex.x = __expf(e.x - mx.x);
    ex.y = __expf(e.y - mx.y);
    ex.z = __expf(e.z - mx.z);
    ex.w = __expf(e.w - mx.w);
    float4 sm = ex;
    #pragma unroll
    for (int msk = 1; msk < 16; msk <<= 1) {
      sm.x += __shfl_xor(sm.x, msk, 32);
      sm.y += __shfl_xor(sm.y, msk, 32);
      sm.z += __shfl_xor(sm.z, msk, 32);
      sm.w += __shfl_xor(sm.w, msk, 32);
    }
    if (lane < 16) {
      s_src[w][j] = sidx;
      s_alpha[w][j * 4 + 0] = ex.x / sm.x;
      s_alpha[w][j * 4 + 1] = ex.y / sm.y;
      s_alpha[w][j * 4 + 2] = ex.z / sm.z;
      s_alpha[w][j * 4 + 3] = ex.w / sm.w;
    }
    asm volatile("s_wait_dscnt 0x0" ::: "memory");   // same-wave LDS RAW fence

    const float* hp = hproj + (size_t)t * NN * HIDN;
    #pragma unroll
    for (int jj = 0; jj < DEG; ++jj) {
      const int s = s_src[w][jj];
      const float a = s_alpha[w][jj * 4 + head];
      const float4* rp = (const float4*)(hp + (size_t)s * HIDN + colb);
      if (jj + 1 < DEG)
        __builtin_prefetch(hp + (size_t)s_src[w][jj + 1] * HIDN + colb, 0, 1);
      float4 r0 = rp[0], r1 = rp[1];
      acc[0] = fmaf(a, r0.x, acc[0]);
      acc[1] = fmaf(a, r0.y, acc[1]);
      acc[2] = fmaf(a, r0.z, acc[2]);
      acc[3] = fmaf(a, r0.w, acc[3]);
      acc[4] = fmaf(a, r1.x, acc[4]);
      acc[5] = fmaf(a, r1.y, acc[5]);
      acc[6] = fmaf(a, r1.z, acc[6]);
      acc[7] = fmaf(a, r1.w, acc[7]);
    }
  }
  const float inv3 = 1.0f / 3.0f;
  #pragma unroll
  for (int i = 0; i < 8; ++i) {
    float bsum = bias[colb + i] + bias[HIDN + colb + i] + bias[2 * HIDN + colb + i];
    float v = (acc[i] + bsum) * inv3;
    if (apply_elu) v = v > 0.f ? v : (__expf(v) - 1.0f);
    outf[(size_t)node * HIDN + colb + i]  = v;
    outbf[(size_t)node * HIDN + colb + i] = (__bf16)v;
  }
}

// ---------------------------------------------------------------------------
// Layer 2 projection (CLS=2, HEADS=1): GEMV + el2/er2 scalars
// ---------------------------------------------------------------------------
__global__ __launch_bounds__(256) void proj2_kernel(const float* __restrict__ hcur,
                                                    const float* __restrict__ W2,
                                                    const float* __restrict__ al2,
                                                    const float* __restrict__ ar2,
                                                    float* __restrict__ h2,
                                                    float* __restrict__ el2,
                                                    float* __restrict__ er2) {
  int id = blockIdx.x * 256 + threadIdx.x;
  if (id >= TT * NN) return;
  int t = id / NN;
  int n = id % NN;
  const float* hr = hcur + (size_t)n * HIDN;
  const float* w  = W2 + (size_t)t * HIDN * 2;
  float c0 = 0.f, c1 = 0.f;
  #pragma unroll 8
  for (int k = 0; k < HIDN; ++k) {
    float hv = hr[k];
    c0 = fmaf(hv, w[2 * k], c0);
    c1 = fmaf(hv, w[2 * k + 1], c1);
  }
  h2[(size_t)id * 2]     = c0;
  h2[(size_t)id * 2 + 1] = c1;
  el2[id] = c0 * al2[t * 2] + c1 * al2[t * 2 + 1];
  er2[id] = c0 * ar2[t * 2] + c1 * ar2[t * 2 + 1];
}

// ---------------------------------------------------------------------------
// Final: per-node 16-edge softmax (1 head), mean over T, bias; softmax over the
// size-1 head axis == 1.0.  out = [logits (N*2) | ones (N*2)]
// ---------------------------------------------------------------------------
__global__ __launch_bounds__(256) void final_kernel(const int* __restrict__ src,
                                                    const float* __restrict__ el2,
                                                    const float* __restrict__ er2,
                                                    const float* __restrict__ h2,
                                                    const float* __restrict__ b2,
                                                    float* __restrict__ out) {
  int n = blockIdx.x * 256 + threadIdx.x;
  if (n >= NN) return;
  float a0 = 0.f, a1 = 0.f;
  for (int t = 0; t < TT; ++t) {
    float erv = er2[(size_t)t * NN + n];
    int sidx[DEG];
    float sc[DEG];
    float mx = -3.4e38f;
    #pragma unroll
    for (int jj = 0; jj < DEG; ++jj) {
      int s = src[(size_t)t * EE + n * DEG + jj];
      sidx[jj] = s;
      float e = lrelu(el2[(size_t)t * NN + s] + erv);
      sc[jj] = e;
      mx = fmaxf(mx, e);
    }
    float sum = 0.f;
    #pragma unroll
    for (int jj = 0; jj < DEG; ++jj) {
      float x = __expf(sc[jj] - mx);
      sc[jj] = x;
      sum += x;
    }
    float inv = 1.0f / sum;
    #pragma unroll
    for (int jj = 0; jj < DEG; ++jj) {
      float a = sc[jj] * inv;
      const float* hv = h2 + ((size_t)t * NN + sidx[jj]) * 2;
      a0 = fmaf(a, hv[0], a0);
      a1 = fmaf(a, hv[1], a1);
    }
  }
  const float inv3 = 1.0f / 3.0f;
  float l0 = (a0 + b2[0] + b2[2] + b2[4]) * inv3;
  float l1 = (a1 + b2[1] + b2[3] + b2[5]) * inv3;
  out[(size_t)n * 2]     = l0;
  out[(size_t)n * 2 + 1] = l1;
  out[(size_t)NN * 2 + n * 2]     = 1.0f;   // softmax over size-1 axis
  out[(size_t)NN * 2 + n * 2 + 1] = 1.0f;
}

// ---------------------------------------------------------------------------
extern "C" void kernel_launch(void* const* d_in, const int* in_sizes, int n_in,
                              void* d_out, int out_size, void* d_ws, size_t ws_size,
                              hipStream_t stream) {
  (void)in_sizes; (void)n_in; (void)out_size; (void)ws_size;
  const float* feat = (const float*)d_in[0];
  const float* W0   = (const float*)d_in[1];
  const float* al0  = (const float*)d_in[2];
  const float* ar0  = (const float*)d_in[3];
  const float* b0   = (const float*)d_in[4];
  const float* W1   = (const float*)d_in[5];
  const float* al1  = (const float*)d_in[6];
  const float* ar1  = (const float*)d_in[7];
  const float* b1   = (const float*)d_in[8];
  const float* W2   = (const float*)d_in[9];
  const float* al2  = (const float*)d_in[10];
  const float* ar2  = (const float*)d_in[11];
  const float* b2   = (const float*)d_in[12];
  const int*   src  = (const int*)d_in[13];
  // d_in[14] (dst) unused: dst is repeat(arange(N), DEG) by construction.
  float* out = (float*)d_out;

  char* p = (char*)d_ws;
  auto take = [&](size_t bytes) { char* r = p; p += (bytes + 255) & ~(size_t)255; return r; };
  __bf16* featbf = (__bf16*)take((size_t)NN * FIN * 2);
  __bf16* hbf    = (__bf16*)take((size_t)NN * HIDN * 2);
  __bf16* wt0    = (__bf16*)take((size_t)TT * HIDN * FIN * 2);
  __bf16* wt1    = (__bf16*)take((size_t)TT * HIDN * HIDN * 2);
  float*  hproj  = (float*)take((size_t)TT * NN * HIDN * 4);
  float*  elb    = (float*)take((size_t)TT * NN * 4 * 4);
  float*  erb    = (float*)take((size_t)TT * NN * 4 * 4);
  float*  hcur   = (float*)take((size_t)NN * HIDN * 4);
  float*  h2     = (float*)take((size_t)TT * NN * 2 * 4);
  float*  el2    = (float*)take((size_t)TT * NN * 4);
  float*  er2    = (float*)take((size_t)TT * NN * 4);

  // --- prep: bf16 activations + transposed bf16 weights
  cvt_feat_kernel<<<(NN * FIN + 255) / 256, 256, 0, stream>>>(feat, featbf, NN * FIN);
  cvt_wt_kernel<<<(TT * FIN * HIDN + 255) / 256, 256, 0, stream>>>(W0, wt0, FIN);
  cvt_wt_kernel<<<(TT * HIDN * HIDN + 255) / 256, 256, 0, stream>>>(W1, wt1, HIDN);

  dim3 ggrid((NN / 16 + 7) / 8, HIDN / 64, TT);

  // --- layer 0
  gemm_wmma<FIN><<<ggrid, 256, 0, stream>>>(featbf, wt0, hproj);
  scores_kernel<<<(TT * NN * HEADS + 255) / 256, 256, 0, stream>>>(hproj, al0, ar0, elb, erb);
  agg_kernel<<<NN / 8, 256, 0, stream>>>(src, elb, erb, hproj, b0, hcur, hbf, 1);

  // --- layer 1
  gemm_wmma<HIDN><<<ggrid, 256, 0, stream>>>(hbf, wt1, hproj);
  scores_kernel<<<(TT * NN * HEADS + 255) / 256, 256, 0, stream>>>(hproj, al1, ar1, elb, erb);
  agg_kernel<<<NN / 8, 256, 0, stream>>>(src, elb, erb, hproj, b1, hcur, hbf, 1);

  // --- layer 2 + output
  proj2_kernel<<<(TT * NN + 255) / 256, 256, 0, stream>>>(hcur, W2, al2, ar2, h2, el2, er2);
  final_kernel<<<(NN + 255) / 256, 256, 0, stream>>>(src, el2, er2, h2, b2, out);
}
